// PixelAttentionModule_40286793237104
// MI455X (gfx1250) — matrix operand order repaired and checked by reading, not compiled, not run on hardware
//
#include <hip/hip_runtime.h>
#include <hip/hip_bf16.h>

typedef __attribute__((ext_vector_type(16))) _Float16 v16h;
typedef __attribute__((ext_vector_type(8)))  _Float16 v8h;
typedef __attribute__((ext_vector_type(8)))  float    v8f;

#define NPIX 9216           // 96*96
#define NBATCH 2
#define NCH 8
#define ROWS_PER_BLOCK 128  // 8 waves * 16 rows
#define BLOCKS_PER_BATCH (NPIX / ROWS_PER_BLOCK)   // 72
#define CSPLIT 4            // column-split factor (occupancy)
#define COLS (NPIX / CSPLIT)                        // 2304 columns per block
#define LOG2E 1.4426950408889634f

// ---------------- Kernel 1: q/k/v 1x1 projections (per-pixel dot over 8 ch) ----
__global__ __launch_bounds__(256) void qkv_proj_kernel(
    const float* __restrict__ x,
    const float* __restrict__ Wq, const float* __restrict__ bq,
    const float* __restrict__ Wk, const float* __restrict__ bk,
    const float* __restrict__ Wv, const float* __restrict__ bv,
    float* __restrict__ q, float* __restrict__ k, float* __restrict__ v)
{
    int i = blockIdx.x * 256 + threadIdx.x;       // 0 .. B*N-1
    if (i >= NBATCH * NPIX) return;
    int b = i / NPIX, n = i - b * NPIX;
    const float* xb = x + (size_t)b * NCH * NPIX + n;
    float aq = bq[0], ak = bk[0], av = bv[0];
#pragma unroll
    for (int c = 0; c < NCH; ++c) {
        float xv = xb[(size_t)c * NPIX];
        aq = fmaf(xv, Wq[c], aq);
        ak = fmaf(xv, Wk[c], ak);
        av = fmaf(xv, Wv[c], av);
    }
    q[i] = aq; k[i] = ak; v[i] = av;
}

// ---------------- Kernel 2: per-batch kmax/kmin (for analytic row-max) --------
__global__ __launch_bounds__(256) void kminmax_kernel(
    const float* __restrict__ k, float* __restrict__ kmm)
{
    __shared__ float smax[256];
    __shared__ float smin[256];
    int b = blockIdx.x, t = threadIdx.x;
    float mx = -3.4e38f, mn = 3.4e38f;
    for (int i = t; i < NPIX; i += 256) {
        float kv = k[b * NPIX + i];
        mx = fmaxf(mx, kv);
        mn = fminf(mn, kv);
    }
    smax[t] = mx; smin[t] = mn;
    __syncthreads();
    for (int s = 128; s > 0; s >>= 1) {
        if (t < s) {
            smax[t] = fmaxf(smax[t], smax[t + s]);
            smin[t] = fminf(smin[t], smin[t + s]);
        }
        __syncthreads();
    }
    if (t == 0) { kmm[b] = smax[0]; kmm[2 + b] = smin[0]; }
}

// Build one 16x32 f16 tile pair: A = exp2(q'*k - m'), B = [v | 1 | 0...].
// kp already includes the lane's hi*8 offset; vp includes hi*8.
__device__ __forceinline__ void build_tile(
    const float* __restrict__ kp, const _Float16* __restrict__ vp,
    float q2, float m2, _Float16 sel_v, _Float16 sel_one,
    v16h& a, v16h& bm)
{
    const float4* kp4 = (const float4*)kp;
    float4 ka = kp4[0];   // K 0..3
    float4 kb = kp4[1];   // K 4..7
    float4 kc = kp4[4];   // K 16..19
    float4 kd = kp4[5];   // K 20..23
    const v8h* vp8 = (const v8h*)vp;
    v8h vlo = vp8[0];
    v8h vhi = vp8[2];

    float kk[16] = {ka.x, ka.y, ka.z, ka.w, kb.x, kb.y, kb.z, kb.w,
                    kc.x, kc.y, kc.z, kc.w, kd.x, kd.y, kd.z, kd.w};
#pragma unroll
    for (int h = 0; h < 16; ++h) {
        // arg <= 0 always; below -126 flushes toward 0 == negligible weight
        float e = __builtin_amdgcn_exp2f(fmaf(q2, kk[h], -m2));
        a[h] = (_Float16)e;
    }
#pragma unroll
    for (int h = 0; h < 8; ++h) {
        // branch-free B build: col0 = v, col1 = 1, others 0
        bm[h]     = (_Float16)(vlo[h] * sel_v + sel_one);
        bm[h + 8] = (_Float16)(vhi[h] * sel_v + sel_one);
    }
}

// ---------------- Kernel 3: rank-1 softmax attention via WMMA -----------------
// grid = (NBATCH*BLOCKS_PER_BATCH, CSPLIT). Each block: 128 rows of one batch,
// one column slice of COLS. Partial (S, Z) per split written to private slices
// of workspace (deterministic; combined in finalize_kernel).
__global__ __launch_bounds__(256) void attn_partial_kernel(
    const float* __restrict__ qv,
    const float* __restrict__ kv,
    const float* __restrict__ vv,
    const float* __restrict__ kmm,
    float* __restrict__ Sp,
    float* __restrict__ Zp)
{
    __shared__ __align__(16) float    s_k[COLS];     // 9 KB
    __shared__ __align__(16) _Float16 s_v[COLS];     // 4.5 KB

    const int tid   = threadIdx.x;
    const int b     = blockIdx.x / BLOCKS_PER_BATCH;
    const int n0    = (blockIdx.x % BLOCKS_PER_BATCH) * ROWS_PER_BLOCK;
    const int split = blockIdx.y;
    const int cbase = split * COLS;

    const float* kb = kv + b * NPIX + cbase;
    const float* vb = vv + b * NPIX + cbase;
    for (int i = tid; i < COLS / 4; i += 256) {
        float4 k4 = ((const float4*)kb)[i];
        ((float4*)s_k)[i] = k4;
        float4 v4 = ((const float4*)vb)[i];
        _Float16* sv = s_v + i * 4;
        sv[0] = (_Float16)v4.x; sv[1] = (_Float16)v4.y;
        sv[2] = (_Float16)v4.z; sv[3] = (_Float16)v4.w;
    }
    __syncthreads();

    const int lane = tid & 31;
    const int wave = tid >> 5;
    const int mrow = lane & 15;     // A-row / C-column index for this lane
    const int hi   = lane >> 4;     // upper half-wave?
    const int row  = n0 + wave * 16 + mrow;   // attention row (within batch)

    const float qn   = qv[b * NPIX + row];
    const float kmax = kmm[b];
    const float kmin = kmm[2 + b];
    const float q2   = qn * LOG2E;
    const float m2   = (qn >= 0.f) ? q2 * kmax : q2 * kmin;  // row max (scaled)

    const int col = mrow;           // which B column this lane supplies
    const _Float16 sel_v   = (col == 0) ? (_Float16)1.0f : (_Float16)0.0f;
    const _Float16 sel_one = (col == 1) ? (_Float16)1.0f : (_Float16)0.0f;

    const int hi8  = hi * 8;
    const int hi16 = hi * 16;

    // two independent accumulators: interleaved exp chains hide TRANS latency
    v8f c0 = {}, c1 = {};
    for (int m0 = 0; m0 < COLS; m0 += 64) {
        v16h a0, b0, a1, b1;
        build_tile(s_k + m0 + hi8,      s_v + m0 + hi8,
                   q2, m2, sel_v, sel_one, a0, b0);
        build_tile(s_k + m0 + 32 + hi8, s_v + m0 + 32 + hi8,
                   q2, m2, sel_v, sel_one, a1, b1);
        c0 = __builtin_amdgcn_wmma_f32_16x16x32_f16(
                 false, a0, false, b0, (short)0, c0, false, false);
        c1 = __builtin_amdgcn_wmma_f32_16x16x32_f16(
                 false, a1, false, b1, (short)0, c1, false, false);
    }

    // C/D layout: lane holds column (lane%16); VGPR r -> M = r + hi*8.
    // Lanes with col==0 hold S partials, col==1 hold Z partials; 8 contiguous
    // rows each -> direct global store into this split's private slice.
    if (col < 2) {
        float* dst = (col == 0 ? Sp : Zp)
                   + ((size_t)split * NBATCH + b) * NPIX + n0 + wave * 16 + hi8;
#pragma unroll
        for (int r = 0; r < 8; ++r) dst[r] = c0[r] + c1[r];
    }
}

// ---------------- Kernel 4: combine splits, divide, residual ------------------
__global__ __launch_bounds__(256) void finalize_kernel(
    const float* __restrict__ x,
    const float* __restrict__ Sp,
    const float* __restrict__ Zp,
    float* __restrict__ out)
{
    int i = blockIdx.x * 256 + threadIdx.x;       // row id 0 .. B*N-1
    if (i >= NBATCH * NPIX) return;
    float S = 0.f, Z = 0.f;
#pragma unroll
    for (int s = 0; s < CSPLIT; ++s) {
        S += Sp[(size_t)s * NBATCH * NPIX + i];
        Z += Zp[(size_t)s * NBATCH * NPIX + i];
    }
    float r = S / Z;
    int b = i / NPIX, n = i - b * NPIX;
    const float* xb = x + (size_t)b * NCH * NPIX + n;
    float*       ob = out + (size_t)b * NCH * NPIX + n;
#pragma unroll
    for (int ch = 0; ch < NCH; ++ch)
        ob[(size_t)ch * NPIX] = r + xb[(size_t)ch * NPIX];
}

extern "C" void kernel_launch(void* const* d_in, const int* in_sizes, int n_in,
                              void* d_out, int out_size, void* d_ws, size_t ws_size,
                              hipStream_t stream) {
    const float* x  = (const float*)d_in[0];
    const float* Wq = (const float*)d_in[1];
    const float* bq = (const float*)d_in[2];
    const float* Wk = (const float*)d_in[3];
    const float* bk = (const float*)d_in[4];
    const float* Wv = (const float*)d_in[5];
    const float* bv = (const float*)d_in[6];
    float* out = (float*)d_out;

    const int BN = NBATCH * NPIX;                 // 18432
    float* ws  = (float*)d_ws;
    float* q   = ws;                              // BN
    float* k   = ws + BN;                         // BN
    float* v   = ws + 2 * BN;                     // BN
    float* kmm = ws + 3 * BN;                     // 4 floats
    float* Sp  = ws + 3 * BN + 16;                // CSPLIT * BN (16B aligned)
    float* Zp  = Sp + CSPLIT * BN;                // CSPLIT * BN

    qkv_proj_kernel<<<(BN + 255) / 256, 256, 0, stream>>>(
        x, Wq, bq, Wk, bk, Wv, bv, q, k, v);
    kminmax_kernel<<<NBATCH, 256, 0, stream>>>(k, kmm);
    dim3 grid(NBATCH * BLOCKS_PER_BATCH, CSPLIT);
    attn_partial_kernel<<<grid, 256, 0, stream>>>(q, k, v, kmm, Sp, Zp);
    finalize_kernel<<<(BN + 255) / 256, 256, 0, stream>>>(x, Sp, Zp, out);
}
